// Retriever_81295140979542
// MI455X (gfx1250) — matrix operand order repaired and checked by reading, not compiled, not run on hardware
//
#include <hip/hip_runtime.h>
#include <hip/hip_bf16.h>

typedef __attribute__((ext_vector_type(16))) __bf16 v16bf;
typedef __attribute__((ext_vector_type(8)))  float  v8f;

#define ROWS    64
#define DIM     384
#define KSTEP   32
#define NSTEPS  (DIM / KSTEP)    // 12
#define MTILES  (ROWS / 16)      // 4
#define WAVES   8
#define TOPK    10
#define AFRAG_ELEMS (MTILES * NSTEPS * 32 * 16)   // 24576 bf16 per table

// ---------------------------------------------------------------------------
// Phase 1: GEMM (bf16x3 split via v_wmma_f32_16x16x32_bf16) + per-wave top-k.
// Each wave owns one 16-column tile at a time and computes ALL 64 rows for it,
// so E (1.54 GB) is streamed from HBM exactly once, with NT (stream) hint.
// ---------------------------------------------------------------------------
__global__ __launch_bounds__(256) void gemm_topk_partial(
    const float* __restrict__ x, const float* __restrict__ E,
    int Nn, int nwavesTotal,
    float* __restrict__ pS, int* __restrict__ pI)
{
    // x pre-split to bf16 hi/lo, pre-swizzled into WMMA A-fragment layout.
    __shared__ __bf16 a_hi[AFRAG_ELEMS];
    __shared__ __bf16 a_lo[AFRAG_ELEMS];
    __shared__ float  stg[WAVES * MTILES * 256];   // 16x16 f32 tile staging

    const int tid = threadIdx.x;

    // ---- Fill A fragments (layout: [m][s][lane][e], e -> K = 16*(lane/16)+e) -
    for (int idx = tid; idx < AFRAG_ELEMS; idx += 256) {
        const int e = idx & 15;
        const int l = (idx >> 4) & 31;
        const int s = (idx >> 9) % NSTEPS;
        const int m = idx / (NSTEPS * 32 * 16);
        const int row  = m * 16 + (l & 15);
        const int koff = (l < 16) ? e : e + 16;
        const float f  = x[row * DIM + s * KSTEP + koff];
        const __bf16 h = (__bf16)f;
        a_hi[idx] = h;
        a_lo[idx] = (__bf16)(f - (float)h);
    }
    __syncthreads();

    const int wave = tid >> 5;
    const int lane = tid & 31;
    const int gw   = blockIdx.x * WAVES + wave;
    const size_t strideN = (size_t)Nn;

    // Two descending-sorted top-10 lists per lane:
    // lanes 0-15 cover mtiles {0,1}, lanes 16-31 cover mtiles {2,3}.
    float ts[2][TOPK];
    int   ti[2][TOPK];
#pragma unroll
    for (int q = 0; q < TOPK; ++q) {
        ts[0][q] = -__builtin_inff(); ts[1][q] = -__builtin_inff();
        ti[0][q] = -1;                ti[1][q] = -1;
    }

    const int T = (Nn + 15) >> 4;  // column tiles
    for (int t = gw; t < T; t += nwavesTotal) {
        const int n0  = t << 4;
        int col = n0 + (lane & 15);
        if (col >= Nn) col = Nn - 1;   // clamp: garbage scores rejected below

        // Running pointer: lane's column, this half-wave's K rows.
        const float* p = E + (size_t)((lane < 16) ? 0 : 16) * strideN + col;

        v8f acc0 = {}, acc1 = {}, acc2 = {}, acc3 = {};

        for (int s = 0; s < NSTEPS; ++s) {
            // ---- B fragment (32x16 bf16): lanes 0-15 K=0..15, 16-31 K=16..31
            float fv[16];
#pragma unroll
            for (int e = 0; e < 16; ++e) {
                fv[e] = __builtin_nontemporal_load(p);  // TH=NT stream
                p += strideN;
            }
            p += 16 * strideN;  // skip the other half-wave's K rows

            v16bf bh, bl;
#pragma unroll
            for (int e = 0; e < 16; ++e) {
                const __bf16 h = (__bf16)fv[e];
                bh[e] = h;
                bl[e] = (__bf16)(fv[e] - (float)h);
            }

            // ---- 4 row tiles x 3 split-WMMAs ------------------------------
#pragma unroll
            for (int m = 0; m < MTILES; ++m) {
                const int off = ((m * NSTEPS + s) * 32 + lane) * 16;
                const v16bf ah = *(const v16bf*)(a_hi + off);  // ds_load_b128
                const v16bf al = *(const v16bf*)(a_lo + off);
                v8f c = (m == 0) ? acc0 : (m == 1) ? acc1 : (m == 2) ? acc2 : acc3;
                c = __builtin_amdgcn_wmma_f32_16x16x32_bf16(false, ah, false, bh,
                                                            (short)0, c, false, false);
                c = __builtin_amdgcn_wmma_f32_16x16x32_bf16(false, ah, false, bl,
                                                            (short)0, c, false, false);
                c = __builtin_amdgcn_wmma_f32_16x16x32_bf16(false, al, false, bh,
                                                            (short)0, c, false, false);
                if (m == 0) acc0 = c; else if (m == 1) acc1 = c;
                else if (m == 2) acc2 = c; else acc3 = c;
            }
        }

        // ---- Stage the 4 tiles through LDS (C/D layout per ISA 7.12.2) ----
        float* st = stg + wave * MTILES * 256;
#pragma unroll
        for (int m = 0; m < MTILES; ++m) {
            const v8f c = (m == 0) ? acc0 : (m == 1) ? acc1 : (m == 2) ? acc2 : acc3;
#pragma unroll
            for (int r = 0; r < 8; ++r) {
                const int rr = (lane < 16) ? r : r + 8;
                st[m * 256 + rr * 16 + (lane & 15)] = c[r];
            }
        }
        asm volatile("s_wait_dscnt 0x0" ::: "memory");  // wave-synchronous

        // ---- Register top-k insertion (unrolled compare-swap, no scratch) -
#pragma unroll
        for (int li = 0; li < 2; ++li) {
            const int m = (lane < 16) ? li : 2 + li;
            const float* rowv = st + m * 256 + (lane & 15) * 16;
            for (int j = 0; j < 16; ++j) {
                const int c2 = n0 + j;
                if (c2 >= Nn) break;
                float v = rowv[j];
                if (v > ts[li][TOPK - 1]) {
                    int vi = c2;
#pragma unroll
                    for (int pp = 0; pp < TOPK; ++pp) {
                        if (v > ts[li][pp]) {
                            float tf = ts[li][pp]; ts[li][pp] = v; v = tf;
                            int   tj = ti[li][pp]; ti[li][pp] = vi; vi = tj;
                        }
                    }
                }
            }
        }
    }

    // ---- Emit per-wave partial top-10 per row -----------------------------
#pragma unroll
    for (int li = 0; li < 2; ++li) {
        const int m    = (lane < 16) ? li : 2 + li;
        const int grow = m * 16 + (lane & 15);
        const size_t base = ((size_t)gw * ROWS + grow) * TOPK;
#pragma unroll
        for (int q = 0; q < TOPK; ++q) {
            pS[base + q] = ts[li][q];
            pI[base + q] = ti[li][q];
        }
    }
}

// ---------------------------------------------------------------------------
// Phase 2: merge per-wave partials -> final top-10 per row.
// One wave per row; lanes scan strided, merge through LDS.
// ---------------------------------------------------------------------------
__global__ __launch_bounds__(32) void topk_reduce(
    const float* __restrict__ pS, const int* __restrict__ pI,
    int nwavesTotal, float* __restrict__ outS, int* __restrict__ outI)
{
    __shared__ float ls[32 * TOPK];
    __shared__ int   lidx[32 * TOPK];

    const int r    = blockIdx.x;
    const int lane = threadIdx.x;

    float ts[TOPK]; int ti[TOPK];
#pragma unroll
    for (int q = 0; q < TOPK; ++q) { ts[q] = -__builtin_inff(); ti[q] = -1; }

    const int C = nwavesTotal * TOPK;
    for (int c = lane; c < C; c += 32) {
        const int w = c / TOPK, q = c % TOPK;
        const size_t o = ((size_t)w * ROWS + r) * TOPK + q;
        float v = pS[o];
        int  id = pI[o];
        if (id >= 0 && v > ts[TOPK - 1]) {
#pragma unroll
            for (int p = 0; p < TOPK; ++p) {
                if (v > ts[p]) {
                    float tf = ts[p]; ts[p] = v; v = tf;
                    int   tj = ti[p]; ti[p] = id; id = tj;
                }
            }
        }
    }
#pragma unroll
    for (int q = 0; q < TOPK; ++q) { ls[lane * TOPK + q] = ts[q]; lidx[lane * TOPK + q] = ti[q]; }
    asm volatile("s_wait_dscnt 0x0" ::: "memory");
    __syncthreads();

    if (lane == 0) {
        float fs[TOPK]; int fi[TOPK];
#pragma unroll
        for (int q = 0; q < TOPK; ++q) { fs[q] = -__builtin_inff(); fi[q] = -1; }
        for (int c = 0; c < 32 * TOPK; ++c) {
            float v = ls[c];
            int  id = lidx[c];
            if (id >= 0 && v > fs[TOPK - 1]) {
#pragma unroll
                for (int p = 0; p < TOPK; ++p) {
                    if (v > fs[p]) {
                        float tf = fs[p]; fs[p] = v; v = tf;
                        int   tj = fi[p]; fi[p] = id; id = tj;
                    }
                }
            }
        }
#pragma unroll
        for (int q = 0; q < TOPK; ++q) {
            outS[r * TOPK + q] = fs[q];
            outI[r * TOPK + q] = fi[q];
        }
    }
}

// ---------------------------------------------------------------------------
extern "C" void kernel_launch(void* const* d_in, const int* in_sizes, int n_in,
                              void* d_out, int out_size, void* d_ws, size_t ws_size,
                              hipStream_t stream)
{
    const float* x = (const float*)d_in[0];
    const float* E = (const float*)d_in[1];
    const int Nn = in_sizes[1] / DIM;       // 1,000,000

    // Partial buffers in workspace: [nw][64][10] scores (f32) then idx (i32).
    const size_t perWaveBytes = (size_t)ROWS * TOPK * (sizeof(float) + sizeof(int));
    long maxBlocks = (long)(ws_size / (perWaveBytes * WAVES));
    int nb = (maxBlocks < 1) ? 1 : (maxBlocks > 128 ? 128 : (int)maxBlocks);
    const int nw = nb * WAVES;

    float* pS = (float*)d_ws;
    int*   pI = (int*)((char*)d_ws + (size_t)nw * ROWS * TOPK * sizeof(float));

    gemm_topk_partial<<<nb, 256, 0, stream>>>(x, E, Nn, nw, pS, pI);
    topk_reduce<<<ROWS, 32, 0, stream>>>(pS, pI, nw,
                                         (float*)d_out, (int*)d_out + ROWS * TOPK);
}
